// Head_43903155699971
// MI455X (gfx1250) — matrix operand order repaired and checked by reading, not compiled, not run on hardware
//
#include <hip/hip_runtime.h>
#include <hip/hip_bf16.h>

// ---------------------------------------------------------------------------
// Causal single-head attention for MI455X (gfx1250), f16 WMMA + flash softmax
// + async-LDS (ASYNCcnt) double-buffered weight staging, sched_group_barrier
// pipelined DS-read/WMMA interleave
// ---------------------------------------------------------------------------

typedef _Float16 v8h  __attribute__((ext_vector_type(8)));
typedef _Float16 v16h __attribute__((ext_vector_type(16)));
typedef float    v8f  __attribute__((ext_vector_type(8)));

#define B_ 4
#define T_ 4096
#define C_ 1024
#define H_ 64

#define WPITCH 40            // LDS row pitch in halves (80B): conflict-free frag reads
#define WROWS  192           // 3 weights * 64 rows
#define WBUF   (WROWS * WPITCH)

static __device__ __forceinline__ v8f wmma_f16(v16h a, v16h b, v8f c) {
    return __builtin_amdgcn_wmma_f32_16x16x32_f16(false, a, false, b, (short)0, c,
                                                  false, false);
}

// gfx1250 async global->LDS copy (tracked by ASYNCcnt), GVS addressing:
// mem = saddr + vaddr(u32); LDS dest = lds_off(u32) per lane.
static __device__ __forceinline__ void async_load_b128(unsigned lds_off,
                                                       unsigned byte_off,
                                                       const void* base) {
    asm volatile("global_load_async_to_lds_b128 %0, %1, %2"
                 :: "v"(lds_off), "v"(byte_off), "s"(base)
                 : "memory");
}

// ---------------------------------------------------------------------------
// Kernel 0: W [C,H] fp32 -> Wt [3][H][C] f16 (transposed so GEMM B-fragments
// are contiguous per lane)
// ---------------------------------------------------------------------------
__global__ void wt_kernel(const float* __restrict__ Wq,
                          const float* __restrict__ Wk,
                          const float* __restrict__ Wv,
                          _Float16* __restrict__ Wt) {
    int idx = blockIdx.x * 256 + threadIdx.x;
    if (idx >= 3 * C_ * H_) return;
    int w = idx / (C_ * H_);
    int r = idx % (C_ * H_);
    int c = r / H_;
    int h = r % H_;
    const float* W = (w == 0) ? Wq : ((w == 1) ? Wk : Wv);
    Wt[(size_t)w * (H_ * C_) + (size_t)h * C_ + c] = (_Float16)W[(size_t)c * H_ + h];
}

// ---------------------------------------------------------------------------
// Kernel 1: fused QKV projection.  One wave = one 16-row tile, all 64 cols,
// all three outputs.  Weight chunks (12 KB per k-step) staged in LDS by the
// whole workgroup with async LDS loads, double-buffered; uniform trip count
// across waves makes workgroup barriers legal here.  sched_group_barrier
// pins: next-A global loads first, then a depth-2 DS-read/WMMA pipeline.
// ---------------------------------------------------------------------------
__global__ __launch_bounds__(256) void qkv_kernel(const float* __restrict__ X,
                                                  const _Float16* __restrict__ Wt,
                                                  _Float16* __restrict__ Qh,
                                                  _Float16* __restrict__ Kh,
                                                  _Float16* __restrict__ Vt) {
    __shared__ __align__(32) _Float16 wlds[2][WBUF];   // 2 x 15360 B

    const int tid  = threadIdx.x;
    const int lane = tid & 31;
    const int wave = tid >> 5;
    const int tile = blockIdx.x * 8 + wave;       // 0 .. B*T/16-1
    const int b    = tile >> 8;                   // 256 tiles per batch
    const int trow = (tile & 255) << 4;
    const int ln   = lane & 15;
    const int hi   = lane >> 4;

    // Cooperative stage of one 3x64x32-half weight chunk: 768 b128 transfers,
    // 3 per thread (one per weight matrix).
    const int srow = tid >> 2;                    // 0..63  (h row)
    const int sseg = tid & 3;                     // 0..3   (8-half segment)
    auto issue_chunk = [&](int kc, int bufsel) {
#pragma unroll
        for (int r = 0; r < 3; ++r) {
            unsigned loff = (unsigned)(uintptr_t)
                &wlds[bufsel][((unsigned)(r * 64 + srow)) * WPITCH + sseg * 8];
            unsigned goff = (unsigned)((r * (H_ * C_) + srow * C_ + kc * 32 + sseg * 8)
                                       * sizeof(_Float16));
            async_load_b128(loff, goff, Wt);
        }
    };

    v8f acc[12];   // [w*4 + t]
#pragma unroll
    for (int i = 0; i < 12; ++i)
        acc[i] = (v8f){0.f, 0.f, 0.f, 0.f, 0.f, 0.f, 0.f, 0.f};

    const float* xrow = X + (size_t)(b * T_ + trow + ln) * C_;
    const int NK = C_ / 32;

    issue_chunk(0, 0);

    // Prologue: A loads for kc = 0
    v8f f0 = *(const v8f*)(xrow + hi * 8);
    v8f f1 = *(const v8f*)(xrow + hi * 8 + 16);

    for (int kc = 0; kc < NK; ++kc) {
        // Fill the other buffer for kc+1 while we compute kc.
        if (kc + 1 < NK) {
            issue_chunk(kc + 1, (kc + 1) & 1);
            asm volatile("s_wait_asynccnt 0x3" ::: "memory");  // kc's 3 ops done
        } else {
            asm volatile("s_wait_asynccnt 0x0" ::: "memory");
        }
        __syncthreads();   // all waves' async writes for kc visible

        // Prefetch future X chunk for this lane's row (streams 64 MB once).
        __builtin_prefetch(xrow + (kc + 3) * 32, 0, 3);

        // A fragment (16x32 f16): convert the loads issued one step ago.
        v16h a;
#pragma unroll
        for (int i = 0; i < 8; ++i) {
            a[i]     = (_Float16)f0[i];
            a[i + 8] = (_Float16)f1[i];
        }
        // Issue next step's A loads now; waited on next iteration.
        if (kc + 1 < NK) {
            const float* pn = xrow + (kc + 1) * 32 + hi * 8;
            f0 = *(const v8f*)pn;
            f1 = *(const v8f*)(pn + 16);
        }

        // B fragments from LDS, depth-2 software pipeline.
        const _Float16* wb = wlds[kc & 1];
        auto load_frag = [&](int idx) {
            const int w = idx >> 2, t = idx & 3;
            const _Float16* bp = wb + (w * 64 + t * 16 + ln) * WPITCH + hi * 8;
            v8h b0 = *(const v8h*)bp;
            v8h b1 = *(const v8h*)(bp + 16);
            v16h bf;
#pragma unroll
            for (int i = 0; i < 8; ++i) { bf[i] = b0[i]; bf[i + 8] = b1[i]; }
            return bf;
        };

        v16h bf0 = load_frag(0);
        v16h bf1 = load_frag(1);
#pragma unroll
        for (int idx = 0; idx < 12; ++idx) {
            acc[idx] = wmma_f16(a, bf0, acc[idx]);
            bf0 = bf1;
            if (idx + 2 < 12) bf1 = load_frag(idx + 2);
        }

        // Pin the schedule: prefetch + 2 next-A VMEM loads first, then
        // 4 DS reads (frag0,frag1), then (WMMA, 2 DS reads) x10, then the
        // last two WMMAs.  Groups are best-effort on the final iteration.
        __builtin_amdgcn_sched_group_barrier(0x020, 3, 0);   // VMEM reads
        __builtin_amdgcn_sched_group_barrier(0x100, 4, 0);   // DS reads
#pragma unroll
        for (int i = 0; i < 10; ++i) {
            __builtin_amdgcn_sched_group_barrier(0x008, 1, 0);   // WMMA
            __builtin_amdgcn_sched_group_barrier(0x100, 2, 0);   // DS reads
        }
        __builtin_amdgcn_sched_group_barrier(0x008, 1, 0);
        __builtin_amdgcn_sched_group_barrier(0x008, 1, 0);

        __syncthreads();   // reads done before this buffer is refilled (kc+2)
    }

    // C/D layout: VGPR v, lane L -> (row = v + hi*8, col = ln)
#pragma unroll
    for (int t = 0; t < 4; ++t) {
#pragma unroll
        for (int v = 0; v < 8; ++v) {
            const int m = v + hi * 8;
            const size_t ro = (size_t)(b * T_ + trow + m) * H_ + t * 16 + ln;
            Qh[ro] = (_Float16)acc[0 * 4 + t][v];
            Kh[ro] = (_Float16)acc[1 * 4 + t][v];
            const int h = t * 16 + ln;
            Vt[(size_t)(b * H_ + h) * T_ + trow + m] = (_Float16)acc[2 * 4 + t][v];
        }
    }
}

// ---------------------------------------------------------------------------
// Kernel 2: flash attention.  One wave = one 16-query tile; key loop in
// chunks of 32 bounded causally.  S via 4 WMMAs/chunk, online softmax with
// shfl reductions, P re-laid out via LDS (wave-private; dscnt wait only —
// waves have different trip counts, so no workgroup barrier), P@V via 4
// WMMAs/chunk with all V fragments loaded up front (claused loads).
// ---------------------------------------------------------------------------
__global__ __launch_bounds__(256) void attn_kernel(const _Float16* __restrict__ Qh,
                                                   const _Float16* __restrict__ Kh,
                                                   const _Float16* __restrict__ Vt,
                                                   float* __restrict__ Out) {
    __shared__ __align__(32) _Float16 pbuf[8][16 * 32];   // per-wave P staging

    const int lane  = threadIdx.x & 31;
    const int wave  = threadIdx.x >> 5;
    const int tile  = blockIdx.x * 8 + wave;
    const int b     = tile >> 8;
    const int qbase = (tile & 255) << 4;
    const int ln    = lane & 15;
    const int hi    = lane >> 4;
    _Float16* pb = pbuf[wave];

    // Q as two A-fragments (d 0..31, 32..63), pre-scaled by 1/sqrt(64)=0.125
    v16h qa0, qa1;
    {
        const _Float16* qr = Qh + (size_t)(b * T_ + qbase + ln) * H_;
        v8h x0 = *(const v8h*)(qr + hi * 16);
        v8h x1 = *(const v8h*)(qr + 8 + hi * 16);
        v8h x2 = *(const v8h*)(qr + 32 + hi * 16);
        v8h x3 = *(const v8h*)(qr + 40 + hi * 16);
        const _Float16 s = (_Float16)0.125f;
#pragma unroll
        for (int i = 0; i < 8; ++i) {
            qa0[i]     = x0[i] * s;
            qa0[i + 8] = x1[i] * s;
            qa1[i]     = x2[i] * s;
            qa1[i + 8] = x3[i] * s;
        }
    }

    v8f o[4];
#pragma unroll
    for (int t = 0; t < 4; ++t) o[t] = (v8f){0.f, 0.f, 0.f, 0.f, 0.f, 0.f, 0.f, 0.f};
    float rmax[8], rsum[8];
#pragma unroll
    for (int v = 0; v < 8; ++v) { rmax[v] = -1e30f; rsum[v] = 0.f; }

    const float L2E = 1.44269504088896341f;
    const int kend = qbase + 16;                  // causal bound (exclusive)

    for (int kbase = 0; kbase < kend; kbase += 32) {
        // Prefetch next chunk's K rows (one 128B row per lane) and V rows.
        const int kn = kbase + 32;
        if (kn < kend) {
            __builtin_prefetch(Kh + (size_t)(b * T_ + kn + lane) * H_, 0, 3);
            __builtin_prefetch(Vt + (size_t)(b * H_ + lane) * T_ + kn, 0, 3);
            __builtin_prefetch(Vt + (size_t)(b * H_ + 32 + lane) * T_ + kn, 0, 3);
        }

        // ---- S = Q * K^T for 32 keys (two 16x16 tiles) ----
        // K B-fragment: lane -> col n = ln (key), k = d = hi*16 + j
        const _Float16* kr = Kh + (size_t)(b * T_ + kbase + ln) * H_ + hi * 16;
        v16h kb0 = *(const v16h*)kr;
        v16h kb1 = *(const v16h*)(kr + 32);
        const _Float16* kr2 = kr + 16 * H_;       // keys kbase+16..kbase+31
        v16h kc0 = *(const v16h*)kr2;
        v16h kc1 = *(const v16h*)(kr2 + 32);

        v8f s0 = (v8f){0.f, 0.f, 0.f, 0.f, 0.f, 0.f, 0.f, 0.f};
        s0 = wmma_f16(qa0, kb0, s0);
        s0 = wmma_f16(qa1, kb1, s0);
        v8f s1 = (v8f){0.f, 0.f, 0.f, 0.f, 0.f, 0.f, 0.f, 0.f};
        s1 = wmma_f16(qa0, kc0, s1);
        s1 = wmma_f16(qa1, kc1, s1);

        // ---- causal mask (only near the diagonal) ----
        if (kbase + 31 > qbase) {
#pragma unroll
            for (int v = 0; v < 8; ++v) {
                const int row = qbase + v + hi * 8;
                if (kbase + ln > row)      s0[v] = -1e30f;
                if (kbase + 16 + ln > row) s1[v] = -1e30f;
            }
        }

        // ---- online softmax (row r lives in VGPR v across a 16-lane half) ----
        float alpha[8];
#pragma unroll
        for (int v = 0; v < 8; ++v) {
            float mx = fmaxf(s0[v], s1[v]);
            mx = fmaxf(mx, __shfl_xor(mx, 1));
            mx = fmaxf(mx, __shfl_xor(mx, 2));
            mx = fmaxf(mx, __shfl_xor(mx, 4));
            mx = fmaxf(mx, __shfl_xor(mx, 8));
            const float mnew = fmaxf(rmax[v], mx);
            alpha[v] = exp2f((rmax[v] - mnew) * L2E);
            rmax[v] = mnew;
            const float p0 = exp2f((s0[v] - mnew) * L2E);
            const float p1 = exp2f((s1[v] - mnew) * L2E);
            float rs = p0 + p1;
            rs += __shfl_xor(rs, 1);
            rs += __shfl_xor(rs, 2);
            rs += __shfl_xor(rs, 4);
            rs += __shfl_xor(rs, 8);
            rsum[v] = rsum[v] * alpha[v] + rs;
            // stage P (C-layout -> LDS at [row][col])
            pb[(v + hi * 8) * 32 + ln]      = (_Float16)p0;
            pb[(v + hi * 8) * 32 + 16 + ln] = (_Float16)p1;
        }

        // wave-private LDS handoff; drain DS ops only (no workgroup barrier)
        asm volatile("s_wait_dscnt 0" ::: "memory");

        // ---- reload P in A-fragment layout (16x32, K = 32 keys) ----
        v16h pa;
        {
            v8h y0 = *(const v8h*)(pb + ln * 32 + hi * 16);
            v8h y1 = *(const v8h*)(pb + ln * 32 + 8 + hi * 16);
#pragma unroll
            for (int i = 0; i < 8; ++i) { pa[i] = y0[i]; pa[i + 8] = y1[i]; }
        }

        // ---- O = O*alpha + P @ V ----
        // All 4 V B-fragments loaded first so the loads clause and the load
        // counter drains progressively across the WMMAs.
        v16h vb[4];
#pragma unroll
        for (int t = 0; t < 4; ++t) {
            const _Float16* vp = Vt + (size_t)(b * H_ + t * 16 + ln) * T_ +
                                 kbase + hi * 16;
            vb[t] = *(const v16h*)vp;
        }
#pragma unroll
        for (int t = 0; t < 4; ++t) {
#pragma unroll
            for (int v = 0; v < 8; ++v) o[t][v] *= alpha[v];
        }
#pragma unroll
        for (int t = 0; t < 4; ++t) o[t] = wmma_f16(pa, vb[t], o[t]);
    }

    // ---- epilogue: O / l, fp32 out [B,T,64] ----
    float inv[8];
#pragma unroll
    for (int v = 0; v < 8; ++v) inv[v] = 1.0f / rsum[v];
#pragma unroll
    for (int t = 0; t < 4; ++t) {
#pragma unroll
        for (int v = 0; v < 8; ++v) {
            const int m = v + hi * 8;
            Out[(size_t)(b * T_ + qbase + m) * H_ + t * 16 + ln] = o[t][v] * inv[v];
        }
    }
}

// ---------------------------------------------------------------------------
extern "C" void kernel_launch(void* const* d_in, const int* in_sizes, int n_in,
                              void* d_out, int out_size, void* d_ws, size_t ws_size,
                              hipStream_t stream) {
    const float* X  = (const float*)d_in[0];
    const float* Wq = (const float*)d_in[1];
    const float* Wk = (const float*)d_in[2];
    const float* Wv = (const float*)d_in[3];

    _Float16* ws = (_Float16*)d_ws;
    _Float16* Wt = ws;                                   // 3*64*1024 halves
    _Float16* Qh = Wt + (size_t)3 * H_ * C_;             // B*T*64 halves
    _Float16* Kh = Qh + (size_t)B_ * T_ * H_;
    _Float16* Vt = Kh + (size_t)B_ * T_ * H_;            // transposed [B,64,T]
    float* Out = (float*)d_out;

    wt_kernel<<<(3 * C_ * H_ + 255) / 256, 256, 0, stream>>>(Wq, Wk, Wv, Wt);
    qkv_kernel<<<(B_ * T_ / 16) / 8, 256, 0, stream>>>(X, Wt, Qh, Kh, Vt);
    attn_kernel<<<(B_ * T_ / 16) / 8, 256, 0, stream>>>(Qh, Kh, Vt, Out);
}
